// SiVALinear_49306224558549
// MI455X (gfx1250) — matrix operand order repaired and checked by reading, not compiled, not run on hardware
//
#include <hip/hip_runtime.h>
#include <hip/hip_bf16.h>

// ---------------- problem constants (from reference) ----------------
#define MTOT 8192      // B*S = 4*2048
#define DIN  4096
#define DOUT 4096
#define DR   512       // decomposition rank (total)
#define TRNK 64        // training rank

// ---------------- tile configuration ----------------
#define BM   128
#define BN   128
#define BK   32
#define LDST 40        // padded LDS row stride in bf16 elems (80B = 20 banks)

typedef __attribute__((ext_vector_type(16))) __bf16 v16bf;
typedef __attribute__((ext_vector_type(8)))  float  v8f;

union FragU { uint4 q[2]; v16bf v; };

// ---------------- CDNA5 async LDS DMA (ASYNCcnt path, ISA 15.18.3 op 98) ----
// LDS byte offset = low 32 bits of the generic pointer to a __shared__ object.
__device__ __forceinline__ void async_copy16(const void* g, void* lds) {
    asm volatile("global_load_async_to_lds_b128 %0, %1, off"
                 :: "v"((unsigned)(unsigned long long)lds), "v"(g)
                 : "memory");
}
__device__ __forceinline__ void wait_async0() {
    asm volatile("s_wait_asynccnt 0x0" ::: "memory");
}

// A-matrix fragment (16x32 bf16, ISA 7.12.2): lane group kgrp=lane>>4 holds
// K = {kgrp*8 .. +7} in VGPRs 0-3 and K = {16+kgrp*8 .. +7} in VGPRs 4-7.
__device__ __forceinline__ v16bf frag_a(const __bf16* base, int row, int kgrp) {
    FragU f;
    const __bf16* p = base + row * LDST;
    f.q[0] = *reinterpret_cast<const uint4*>(p + kgrp * 8);
    f.q[1] = *reinterpret_cast<const uint4*>(p + 16 + kgrp * 8);
    return f.v;
}
// B-matrix fragment (32x16 bf16): lane group kgrp holds K = {kgrp*8 .. +7}
// in VGPRs 0-3 and K = {16+kgrp*8 .. +7} in VGPRs 4-7, column N = lane&15.
__device__ __forceinline__ v16bf frag_b(const __bf16* base, int row, int kgrp) {
    FragU f;
    const __bf16* p = base + row * LDST;
    f.q[0] = *reinterpret_cast<const uint4*>(p + kgrp * 8);
    f.q[1] = *reinterpret_cast<const uint4*>(p + 16 + kgrp * 8);
    return f.v;
}

// async-stage a [128 rows x 32 bf16] tile (row-major, leading dim ldk) into LDS
__device__ __forceinline__ void stage_tile_async(const __bf16* __restrict__ G,
                                                 size_t ldk, int rbase, int kk,
                                                 int t, __bf16* lds) {
#pragma unroll
    for (int p = 0; p < 2; ++p) {
        int row = p * 64 + (t >> 2);
        int kq  = (t & 3) * 8;
        async_copy16(G + (size_t)(rbase + row) * ldk + kk + kq,
                     &lds[row * LDST + kq]);
    }
}

// ---------------- prep: pack factors to bf16, fold s into V rows ------------
__global__ void siva_prep_v(const float* __restrict__ s,
                            const float* __restrict__ v_t,
                            const float* __restrict__ v_d,
                            __bf16* __restrict__ Vb) {
    int idx = blockIdx.x * blockDim.x + threadIdx.x;
    if (idx >= DR * DIN) return;
    int r = idx >> 12;
    int i = idx & (DIN - 1);
    float val = (r < TRNK) ? v_t[r * DIN + i] : v_d[(r - TRNK) * DIN + i];
    Vb[idx] = (__bf16)(s[r] * val);
}

__global__ void siva_prep_u(const float* __restrict__ u_t,
                            const float* __restrict__ u_d,
                            __bf16* __restrict__ Ub) {
    int idx = blockIdx.x * blockDim.x + threadIdx.x;
    if (idx >= DOUT * DR) return;
    int o = idx >> 9;
    int r = idx & (DR - 1);
    float val = (r < TRNK) ? u_t[o * TRNK + r] : u_d[o * (DR - TRNK) + (r - TRNK)];
    Ub[idx] = (__bf16)val;
}

// ---------------- GEMM1: T[m][n] = sum_k x[m][k] * Vb[n][k] -----------------
// M=MTOT, N=DR, K=DIN. x fp32 -> bf16 on the fly; double-buffered LDS;
// x staged via registers (needs cvt), Vb staged via async LDS DMA.
__global__ __launch_bounds__(256)
void siva_gemm1(const float* __restrict__ X,
                const __bf16* __restrict__ Vb,
                __bf16* __restrict__ T) {
    __shared__ __bf16 As[2][BM * LDST];
    __shared__ __bf16 Bs[2][BN * LDST];

    const int bm = blockIdx.x * BM;
    const int bn = blockIdx.y * BN;
    const int t    = threadIdx.x;
    const int lane = t & 31;
    const int wave = t >> 5;
    const int wm = (wave & 3) * 32;   // 4 waves along M
    const int wn = (wave >> 2) * 64;  // 2 waves along N
    const int laneM = lane & 15;
    const int kgrp  = lane >> 4;
    const int NK = DIN / BK;

    const int arow = t >> 3;          // A staging: 32 rows/pass, 4 passes
    const int akq  = (t & 7) * 4;

    v8f acc[2][4];
#pragma unroll
    for (int i = 0; i < 2; ++i)
#pragma unroll
        for (int j = 0; j < 4; ++j) { v8f z = {}; acc[i][j] = z; }

    float4 aReg[4];
    // prologue: issue slab 0
#pragma unroll
    for (int p = 0; p < 4; ++p)
        aReg[p] = *reinterpret_cast<const float4*>(
            X + (size_t)(bm + p * 32 + arow) * DIN + akq);
    stage_tile_async(Vb, DIN, bn, 0, t, Bs[0]);

    for (int k = 0; k < NK; ++k) {
        const int cur = k & 1;
        // commit A slab k: cvt f32->bf16, store to LDS (loads issued 1 iter ago)
#pragma unroll
        for (int p = 0; p < 4; ++p) {
            __bf16* dst = &As[cur][(p * 32 + arow) * LDST + akq];
            dst[0] = (__bf16)aReg[p].x; dst[1] = (__bf16)aReg[p].y;
            dst[2] = (__bf16)aReg[p].z; dst[3] = (__bf16)aReg[p].w;
        }
        wait_async0();        // B tile DMA for buf[cur] complete (my wave)
        __syncthreads();      // all waves' stores visible; prev readers done

        if (k + 1 < NK) {     // issue slab k+1 into the other buffer
            const int kk = (k + 1) * BK;
#pragma unroll
            for (int p = 0; p < 4; ++p)
                aReg[p] = *reinterpret_cast<const float4*>(
                    X + (size_t)(bm + p * 32 + arow) * DIN + kk + akq);
            stage_tile_async(Vb, DIN, bn, kk, t, Bs[cur ^ 1]);
            if (k + 2 < NK)   // prefetch slab k+2 of x (global_prefetch_b8)
                __builtin_prefetch(X + (size_t)(bm + arow) * DIN + kk + BK, 0, 0);
        }

        v16bf af[2], bf[4];
#pragma unroll
        for (int i = 0; i < 2; ++i) af[i] = frag_a(As[cur], wm + i * 16 + laneM, kgrp);
#pragma unroll
        for (int j = 0; j < 4; ++j) bf[j] = frag_b(Bs[cur], wn + j * 16 + laneM, kgrp);
#pragma unroll
        for (int i = 0; i < 2; ++i)
#pragma unroll
            for (int j = 0; j < 4; ++j)
                acc[i][j] = __builtin_amdgcn_wmma_f32_16x16x32_bf16(
                    false, af[i], false, bf[j], (short)0, acc[i][j], false, false);
    }

    // epilogue: D layout -> row = r + kgrp*8, col = lane&15
#pragma unroll
    for (int i = 0; i < 2; ++i)
#pragma unroll
        for (int j = 0; j < 4; ++j) {
            int col = bn + wn + j * 16 + laneM;
#pragma unroll
            for (int r = 0; r < 8; ++r) {
                int row = bm + wm + i * 16 + kgrp * 8 + r;
                T[(size_t)row * DR + col] = (__bf16)acc[i][j][r];
            }
        }
}

// ---------------- GEMM2: Y[m][o] = sum_r T[m][r]*Ub[o][r] + bias[o] ---------
// M=MTOT, N=DOUT, K=DR. Both operands bf16: pure async LDS DMA staging,
// double-buffered, one barrier per K-step.
__global__ __launch_bounds__(256)
void siva_gemm2(const __bf16* __restrict__ T,
                const __bf16* __restrict__ Ub,
                const float* __restrict__ bias,
                float* __restrict__ Y) {
    __shared__ __bf16 As[2][BM * LDST];
    __shared__ __bf16 Bs[2][BN * LDST];

    const int bm = blockIdx.x * BM;
    const int bn = blockIdx.y * BN;
    const int t    = threadIdx.x;
    const int lane = t & 31;
    const int wave = t >> 5;
    const int wm = (wave & 3) * 32;
    const int wn = (wave >> 2) * 64;
    const int laneM = lane & 15;
    const int kgrp  = lane >> 4;
    const int NK = DR / BK;

    v8f acc[2][4];
#pragma unroll
    for (int i = 0; i < 2; ++i)
#pragma unroll
        for (int j = 0; j < 4; ++j) { v8f z = {}; acc[i][j] = z; }

    // prologue: issue slab 0 DMA
    stage_tile_async(T,  DR, bm, 0, t, As[0]);
    stage_tile_async(Ub, DR, bn, 0, t, Bs[0]);

    for (int k = 0; k < NK; ++k) {
        const int cur = k & 1;
        wait_async0();        // my DMA into buf[cur] complete
        __syncthreads();      // everyone's complete; prev readers done

        if (k + 1 < NK) {
            const int kk = (k + 1) * BK;
            stage_tile_async(T,  DR, bm, kk, t, As[cur ^ 1]);
            stage_tile_async(Ub, DR, bn, kk, t, Bs[cur ^ 1]);
        }

        v16bf af[2], bf[4];
#pragma unroll
        for (int i = 0; i < 2; ++i) af[i] = frag_a(As[cur], wm + i * 16 + laneM, kgrp);
#pragma unroll
        for (int j = 0; j < 4; ++j) bf[j] = frag_b(Bs[cur], wn + j * 16 + laneM, kgrp);
#pragma unroll
        for (int i = 0; i < 2; ++i)
#pragma unroll
            for (int j = 0; j < 4; ++j)
                acc[i][j] = __builtin_amdgcn_wmma_f32_16x16x32_bf16(
                    false, af[i], false, bf[j], (short)0, acc[i][j], false, false);
    }

#pragma unroll
    for (int i = 0; i < 2; ++i)
#pragma unroll
        for (int j = 0; j < 4; ++j) {
            int col = bn + wn + j * 16 + laneM;
            float bj = bias[col];
#pragma unroll
            for (int r = 0; r < 8; ++r) {
                int row = bm + wm + i * 16 + kgrp * 8 + r;
                Y[(size_t)row * DOUT + col] = acc[i][j][r] + bj;
            }
        }
}

// ---------------- host launcher ----------------
extern "C" void kernel_launch(void* const* d_in, const int* in_sizes, int n_in,
                              void* d_out, int out_size, void* d_ws, size_t ws_size,
                              hipStream_t stream) {
    const float* x    = (const float*)d_in[0];
    const float* u_t  = (const float*)d_in[1];
    const float* s    = (const float*)d_in[2];
    const float* v_t  = (const float*)d_in[3];
    const float* u_d  = (const float*)d_in[4];
    const float* v_d  = (const float*)d_in[5];
    const float* bias = (const float*)d_in[6];
    float* y = (float*)d_out;

    // workspace layout (16 MB total; L2-resident on MI455X)
    __bf16* Vb = (__bf16*)d_ws;                    // [DR   x DIN ] scaled V
    __bf16* Ub = Vb + (size_t)DR * DIN;            // [DOUT x DR  ]
    __bf16* T  = Ub + (size_t)DOUT * DR;           // [MTOT x DR  ] intermediate

    siva_prep_v<<<(DR * DIN + 255) / 256, 256, 0, stream>>>(s, v_t, v_d, Vb);
    siva_prep_u<<<(DOUT * DR + 255) / 256, 256, 0, stream>>>(u_t, u_d, Ub);

    dim3 g1(MTOT / BM, DR / BN);     // 64 x 4
    siva_gemm1<<<g1, 256, 0, stream>>>(x, Vb, T);

    dim3 g2(MTOT / BM, DOUT / BN);   // 64 x 32
    siva_gemm2<<<g2, 256, 0, stream>>>(T, Ub, bias, y);

    (void)in_sizes; (void)n_in; (void)out_size; (void)ws_size;
}
